// BertAmir4_83262236000547
// MI455X (gfx1250) — matrix operand order repaired and checked by reading, not compiled, not run on hardware
//
#include <hip/hip_runtime.h>
#include <hip/hip_bf16.h>
#include <math.h>

// ---------------- problem constants ----------------
#define BB   64     // batch
#define TT   128    // seq len
#define LL   160
#define DIN  3072
#define HH   300
#define G4   1200   // 4*H
#define H2   600    // 2*H
#define PP   34

typedef __attribute__((ext_vector_type(16))) __bf16        v16bf;
typedef __attribute__((ext_vector_type(8)))  float         v8f;
typedef __attribute__((ext_vector_type(8)))  unsigned int  v8u;
typedef __attribute__((ext_vector_type(4)))  unsigned int  v4u;
typedef __bf16 bf16_t;

static __device__ __forceinline__ __bf16 f2bf(float f) {
  unsigned int u = __builtin_bit_cast(unsigned int, f);
  unsigned int r = (u + 0x7FFFu + ((u >> 16) & 1u)) >> 16;
  unsigned short s = (unsigned short)r;
  return __builtin_bit_cast(__bf16, s);
}

static __device__ __forceinline__ __bf16 bfzero() {
  return __builtin_bit_cast(__bf16, (unsigned short)0);
}

static __device__ __forceinline__ float sigmoidf_(float x) {
  return 1.0f / (1.0f + __expf(-x));
}

static __device__ __forceinline__ v8f wmma_bf16(v16bf a, v16bf b, v8f c) {
  return __builtin_amdgcn_wmma_f32_16x16x32_bf16(
      false, a, false, b, (short)0, c, false, false);
}

// ------------------------------------------------------------------
// Batched GEMM via V_WMMA_F32_16X16X32_BF16 with block-cooperative
// LDS staging of B.
//   block = 256 threads = 8 waves -> 128x32 C macro-tile
//   per 32-K chunk: B(32x32) staged once into LDS (transposed, padded
//   row stride 40 halves), each wave reads fragments as ds b128 loads.
// All N/K edges handled in staging (zero-fill), so the compute path is
// branch-free; waves whose rows exceed M clamp their A rows (M%16==0
// for every call site) and skip the store. EXEC is all-ones at every
// WMMA; barriers are reached by all waves.
// epi: x = acc; x += addm[b][m][n]; x /= rdiv[b][m]; x += bias[n];
//      act (0 none,1 relu,2 sigmoid); x *= cscale[b][n];
//      optional fp32 store (C) and/or bf16 store (Cbf).
// ------------------------------------------------------------------
__global__ void gemm_wmma(
    const bf16_t* __restrict__ A, long long sA, int lda,
    const bf16_t* __restrict__ B, long long sB, int ldb,
    float* __restrict__ C, bf16_t* __restrict__ Cbf, long long sC, int ldc,
    const float* __restrict__ bias,
    const float* __restrict__ addm, long long sAdd, int ldadd,
    const float* __restrict__ rdiv, long long sDiv,
    const float* __restrict__ cscale, long long sScale,
    int M, int N, int K, int nbatch, int act)
{
  __shared__ bf16_t Bs[32 * 40];     // [n][k], padded stride vs bank conflicts

  const int tid  = threadIdx.x;
  const int lane = tid & 31;
  const int wave = tid >> 5;

  const int tm  = (M + 15) >> 4;     // 16-row tiles (M % 16 == 0 at call sites)
  const int tmB = (tm + 7) >> 3;     // 128-row macro tiles
  const int tn2 = (N + 31) >> 5;     // 32-wide column tiles

  long long blk = blockIdx.x;
  const long long perBatch = (long long)tmB * tn2;
  int batch = (int)(blk / perBatch);
  long long rem = blk - (long long)batch * perBatch;
  int mb = (int)(rem / tn2);
  int nt = (int)(rem - (long long)mb * tn2);

  const int row0  = (mb * 8 + wave) << 4;
  const int row0c = (row0 + 16 <= M) ? row0 : (M - 16);  // clamp surplus waves
  const int col0  = nt << 5;

  const bf16_t* Ab = A + (long long)batch * sA;
  const bf16_t* Bb = B + (long long)batch * sB;

  const int mA = row0c + (lane & 15);
  const int hi = lane >> 4;          // lane half selects K sub-ranges
  const int nl = lane & 15;

  const unsigned int* Arow32 = (const unsigned int*)(Ab + (long long)mA * lda);

  // staging coords: thread covers 4 consecutive n at one k row
  const int kr = tid >> 3;           // 0..31
  const int nc = (tid & 7) << 2;     // 0,4,...,28
  const bool colFull = (col0 + 32 <= N);

  v8f acc0 = {};
  v8f acc1 = {};

  const int kIters = (K + 31) >> 5;
  for (int kk = 0; kk < kIters; ++kk) {
    const int k0 = kk << 5;

    __syncthreads();
    // ---- stage B chunk (transposed) into LDS ----
    const int kg = k0 + kr;
    if (colFull && kg < K) {
      uint2 q = *(const uint2*)(Bb + (long long)kg * ldb + col0 + nc);
      Bs[(nc + 0) * 40 + kr] = __builtin_bit_cast(bf16_t, (unsigned short)(q.x & 0xffffu));
      Bs[(nc + 1) * 40 + kr] = __builtin_bit_cast(bf16_t, (unsigned short)(q.x >> 16));
      Bs[(nc + 2) * 40 + kr] = __builtin_bit_cast(bf16_t, (unsigned short)(q.y & 0xffffu));
      Bs[(nc + 3) * 40 + kr] = __builtin_bit_cast(bf16_t, (unsigned short)(q.y >> 16));
    } else {
      #pragma unroll
      for (int j = 0; j < 4; ++j) {
        int n = col0 + nc + j;
        bf16_t v = (kg < K && n < N) ? Bb[(long long)kg * ldb + n] : bfzero();
        Bs[(nc + j) * 40 + kr] = v;
      }
    }
    __syncthreads();

    // ---- A fragment: 8 x b32 from global (bf16 pairs (k,k+1)) ----
    // K-tail overread is safe: B is zero-staged for k >= K, and the bf16
    // arena is padded, so garbage lanes contribute exact zeros.
    v8u au;
    #pragma unroll
    for (int p = 0; p < 8; ++p) {
      int offHalf = (hi << 3) + p;
      au[p] = Arow32[(k0 >> 1) + offHalf];
    }
    v16bf a = __builtin_bit_cast(v16bf, au);
    if (kk + 1 < kIters)
      __builtin_prefetch(Arow32 + ((k0 + 32) >> 1), 0, 3);

    // ---- B fragments from LDS: 16 contiguous halves each ----
    const v4u* p0 = (const v4u*)(&Bs[nl * 40 + (hi << 4)]);
    const v4u* p1 = (const v4u*)(&Bs[(16 + nl) * 40 + (hi << 4)]);
    v4u b0lo = p0[0], b0hi = p0[1];
    v4u b1lo = p1[0], b1hi = p1[1];
    v8u bu0, bu1;
    #pragma unroll
    for (int j = 0; j < 4; ++j) {
      bu0[j] = b0lo[j]; bu0[4 + j] = b0hi[j];
      bu1[j] = b1lo[j]; bu1[4 + j] = b1hi[j];
    }
    v16bf b0 = __builtin_bit_cast(v16bf, bu0);
    v16bf b1 = __builtin_bit_cast(v16bf, bu1);

    acc0 = wmma_bf16(a, b0, acc0);
    acc1 = wmma_bf16(a, b1, acc1);
  }

  // ---------------- epilogue + store (both column tiles) ----------------
  if (row0 < M) {
    #pragma unroll
    for (int half = 0; half < 2; ++half) {
      const int n = col0 + nl + (half << 4);
      v8f acc = half ? acc1 : acc0;
      if (n < N) {
        #pragma unroll
        for (int v = 0; v < 8; ++v) {
          int m = row0 + v + (hi << 3);
          if (m < M) {
            float x = acc[v];
            if (addm)   x += addm[(long long)batch * sAdd + (long long)m * ldadd + n];
            if (rdiv)   x /= rdiv[(long long)batch * sDiv + m];
            if (bias)   x += bias[n];
            if (act == 1) x = fmaxf(x, 0.0f);
            else if (act == 2) x = sigmoidf_(x);
            if (cscale) x *= cscale[(long long)batch * sScale + n];
            long long o = (long long)batch * sC + (long long)m * ldc + n;
            if (C)   C[o]   = x;
            if (Cbf) Cbf[o] = f2bf(x);
          }
        }
      }
    }
  }
}

// ------------------- small helper kernels -------------------------

// fp32 -> bf16 staging pass
__global__ void to_bf16(const float* __restrict__ src, bf16_t* __restrict__ dst,
                        long long n) {
  long long i = (long long)blockIdx.x * blockDim.x + threadIdx.x;
  if (i < n) dst[i] = f2bf(src[i]);
}

__global__ void zero_f32(float* p, int n) {
  int i = blockIdx.x * blockDim.x + threadIdx.x;
  if (i < n) p[i] = 0.0f;
}

__global__ void zero_bf(bf16_t* p, int n) {
  int i = blockIdx.x * blockDim.x + threadIdx.x;
  if (i < n) p[i] = bfzero();
}

// LSTM cell elementwise update for both directions at step t
// state layout (floats): c_f:38400, c_b:57600, gates_f:76800, gates_b:153600
// h kept only as bf16 (h_bf[2][B*H]) since it feeds the next step GEMM.
__global__ void lstm_cell(float* __restrict__ state,
                          bf16_t* __restrict__ h_bf,
                          float* __restrict__ xcat,
                          bf16_t* __restrict__ xcat_bf, int t) {
  int idx = blockIdx.x * blockDim.x + threadIdx.x;
  if (idx >= 2 * BB * HH) return;
  int dir = idx / (BB * HH);
  int r   = idx - dir * (BB * HH);
  int b   = r / HH;
  int hc  = r - b * HH;
  float* c = state + 2 * BB * HH + dir * (BB * HH);
  const float* g = state + 4 * BB * HH + dir * (BB * G4) + b * G4;
  float ig = sigmoidf_(g[hc]);
  float fg = sigmoidf_(g[HH  + hc]);
  float gg = tanhf   (g[2*HH + hc]);
  float og = sigmoidf_(g[3*HH + hc]);
  float cv = fg * c[r] + ig * gg;
  float hv = og * tanhf(cv);
  c[r] = cv;
  h_bf[idx] = f2bf(hv);
  int tw = (dir == 0) ? t : (TT - 1 - t);
  long long o = ((long long)b * TT + tw) * H2 + dir * HH + hc;
  xcat[o]    = hv;
  xcat_bf[o] = f2bf(hv);
}

// aspect[b,e] = xcat[b, anchor[b], e]  (fp32 + bf16 copies)
__global__ void aspect_gather(const float* __restrict__ xcat,
                              const int* __restrict__ anchor,
                              float* __restrict__ aspect,
                              bf16_t* __restrict__ aspect_bf) {
  int i = blockIdx.x * blockDim.x + threadIdx.x;
  if (i >= BB * H2) return;
  int b = i / H2, e = i - b * H2;
  float v = xcat[((long long)b * TT + anchor[b]) * H2 + e];
  aspect[i]    = v;
  aspect_bf[i] = f2bf(v);
}

// denom[b*T+t] = 1 + sum_s adj[b,t,s]
__global__ void row_degree(const float* __restrict__ adj, float* __restrict__ denom) {
  int i = blockIdx.x * blockDim.x + threadIdx.x;
  if (i >= BB * TT) return;
  const float* r = adj + (long long)i * TT;
  float s = 1.0f;
  for (int j = 0; j < TT; ++j) s += r[j];
  denom[i] = s;
}

// dst[b,e] = (scale?scale[b,e]:1) * max_t src[b,t,e]   (scale>0 -> factoring valid)
__global__ void max_over_t(const float* __restrict__ src,
                           const float* __restrict__ scale,
                           float* __restrict__ dst) {
  int i = blockIdx.x * blockDim.x + threadIdx.x;
  if (i >= BB * H2) return;
  int b = i / H2, e = i - b * H2;
  float m = -3.4e38f;
  const float* p = src + (long long)b * TT * H2 + e;
  for (int t = 0; t < TT; ++t) m = fmaxf(m, p[(long long)t * H2]);
  if (scale) m *= scale[i];
  dst[i] = m;
}

// cat2[b, 0:600]=aspect, [600:1200]=omax   (bf16, GEMM operand only)
__global__ void concat2(const float* __restrict__ aspect,
                        const float* __restrict__ omax,
                        bf16_t* __restrict__ cat2) {
  int i = blockIdx.x * blockDim.x + threadIdx.x;
  if (i >= BB * 2 * H2) return;
  int b = i / (2 * H2), j = i - b * (2 * H2);
  float v = (j < H2) ? aspect[b * H2 + j] : omax[b * H2 + (j - H2)];
  cat2[i] = f2bf(v);
}

// fccat[(b*T+t), 0:600]=x2[b,t], [600:1200]=aspect[b]   (bf16)
__global__ void concat_fc(const float* __restrict__ x2,
                          const float* __restrict__ aspect,
                          bf16_t* __restrict__ fccat) {
  long long i = (long long)blockIdx.x * blockDim.x + threadIdx.x;
  if (i >= (long long)BB * TT * 2 * H2) return;
  int row = (int)(i / (2 * H2));
  int j = (int)(i - (long long)row * (2 * H2));
  int b = row / TT;
  float v = (j < H2) ? x2[(long long)row * H2 + j] : aspect[b * H2 + (j - H2)];
  fccat[i] = f2bf(v);
}

// scores[b,t] = sum_p logits[b,p] * outw[b,t,p]
__global__ void score_dot(const float* __restrict__ logits,
                          const float* __restrict__ outw,
                          float* __restrict__ scores) {
  int i = blockIdx.x * blockDim.x + threadIdx.x;
  if (i >= BB * TT) return;
  int b = i / TT;
  float s = 0.0f;
  const float* ow = outw + (long long)i * PP;
  const float* lg = logits + b * PP;
  for (int p = 0; p < PP; ++p) s += lg[p] * ow[p];
  scores[i] = s;
}

// xy = (1/B) * sum_{b,e} x1*y1   -> out[2176]
__global__ void xy_reduce(const float* __restrict__ x1,
                          const float* __restrict__ y1,
                          float* __restrict__ out) {
  __shared__ float sh[256];
  float loc = 0.0f;
  for (int i = threadIdx.x; i < BB * H2; i += 256) loc += x1[i] * y1[i];
  sh[threadIdx.x] = loc; __syncthreads();
  for (int o = 128; o > 0; o >>= 1) {
    if (threadIdx.x < o) sh[threadIdx.x] += sh[threadIdx.x + o];
    __syncthreads();
  }
  if (threadIdx.x == 0) out[BB * PP] = sh[0] / (float)BB;
}

// per-batch softmax(scores) . softmax(dist) -> bdot[b]; 128 threads/block
__global__ void kl_batch(const float* __restrict__ scores,
                         const float* __restrict__ dist,
                         float* __restrict__ bdot) {
  __shared__ float sh[TT];
  int b = blockIdx.x, t = threadIdx.x;

  float s = scores[b * TT + t];
  sh[t] = s; __syncthreads();
  for (int o = 64; o > 0; o >>= 1) { if (t < o) sh[t] = fmaxf(sh[t], sh[t + o]); __syncthreads(); }
  float mx = sh[0]; __syncthreads();
  float e1 = __expf(s - mx);
  sh[t] = e1; __syncthreads();
  for (int o = 64; o > 0; o >>= 1) { if (t < o) sh[t] += sh[t + o]; __syncthreads(); }
  float p1 = e1 / sh[0]; __syncthreads();

  float d = dist[b * TT + t];
  sh[t] = d; __syncthreads();
  for (int o = 64; o > 0; o >>= 1) { if (t < o) sh[t] = fmaxf(sh[t], sh[t + o]); __syncthreads(); }
  float mx2 = sh[0]; __syncthreads();
  float e2 = __expf(d - mx2);
  sh[t] = e2; __syncthreads();
  for (int o = 64; o > 0; o >>= 1) { if (t < o) sh[t] += sh[t + o]; __syncthreads(); }
  float p2 = e2 / sh[0]; __syncthreads();

  sh[t] = p1 * p2; __syncthreads();
  for (int o = 64; o > 0; o >>= 1) { if (t < o) sh[t] += sh[t + o]; __syncthreads(); }
  if (t == 0) bdot[b] = sh[0];
}

__global__ void finalize_kl(const float* __restrict__ bdot, float* __restrict__ out) {
  if (threadIdx.x == 0 && blockIdx.x == 0) {
    float s = 0.0f;
    for (int i = 0; i < BB; ++i) s += bdot[i];
    out[BB * PP + 1] = s / (float)BB;
  }
}

// ------------------------------------------------------------------
extern "C" void kernel_launch(void* const* d_in, const int* in_sizes, int n_in,
                              void* d_out, int out_size, void* d_ws, size_t ws_size,
                              hipStream_t stream) {
  (void)in_sizes; (void)n_in; (void)out_size; (void)ws_size;

  const float* x_bert    = (const float*)d_in[0];
  const float* transform = (const float*)d_in[1];
  const float* dist      = (const float*)d_in[2];
  const float* adj       = (const float*)d_in[3];
  const float* Wih_f = (const float*)d_in[4];
  const float* Whh_f = (const float*)d_in[5];
  const float* b_f   = (const float*)d_in[6];
  const float* Wih_b = (const float*)d_in[7];
  const float* Whh_b = (const float*)d_in[8];
  const float* b_b   = (const float*)d_in[9];
  const float* gc1_w = (const float*)d_in[10];
  const float* gc1_b = (const float*)d_in[11];
  const float* gc2_w = (const float*)d_in[12];
  const float* gc2_b = (const float*)d_in[13];
  const float* gw[2][3] = {
    { (const float*)d_in[14], (const float*)d_in[16], (const float*)d_in[18] },
    { (const float*)d_in[20], (const float*)d_in[22], (const float*)d_in[24] } };
  const float* gb[2][3] = {
    { (const float*)d_in[15], (const float*)d_in[17], (const float*)d_in[19] },
    { (const float*)d_in[21], (const float*)d_in[23], (const float*)d_in[25] } };
  const float* dense_w = (const float*)d_in[26];
  const float* dense_b = (const float*)d_in[27];
  const float* fc_w    = (const float*)d_in[28];
  const float* fc_b    = (const float*)d_in[29];
  const int*   anchor  = (const int*)d_in[30];

  float* out = (float*)d_out;
  float* ws  = (float*)d_ws;

  // ---------------- fp32 workspace (float offsets) ----------------
  long long fo = 0;
  auto falloc = [&](long long n) { long long o = fo; fo += n; return o; };
  const long long o_pre_f = falloc((long long)BB*TT*G4);
  const long long o_pre_b = falloc((long long)BB*TT*G4);
  const long long o_xcat  = falloc((long long)BB*TT*H2);
  const long long o_state = falloc(2*(BB*HH)*2 + 2*(BB*G4));
  const long long o_asp   = falloc(BB*H2);
  const long long o_g1    = falloc(BB*H2);
  const long long o_g2    = falloc(BB*H2);
  const long long o_gcn1  = falloc((long long)BB*TT*H2);
  const long long o_x2    = falloc((long long)BB*TT*H2);
  const long long o_den   = falloc(BB*TT);
  const long long o_x1    = falloc(BB*H2);
  const long long o_y1    = falloc(BB*H2);
  const long long o_omax  = falloc(BB*H2);
  const long long o_outw  = falloc((long long)BB*TT*PP);
  const long long o_scr   = falloc(BB*TT);
  const long long o_bdot  = falloc(64);

  float* pre_f = ws + o_pre_f;
  float* pre_b = ws + o_pre_b;
  float* xcat  = ws + o_xcat;
  float* state = ws + o_state;
  float* gts_f = state + 4*BB*HH;
  float* gts_b = state + 4*BB*HH + BB*G4;
  float* asp   = ws + o_asp;
  float* g1    = ws + o_g1;
  float* g2    = ws + o_g2;
  float* gcn1  = ws + o_gcn1;
  float* x2    = ws + o_x2;
  float* den   = ws + o_den;
  float* x1    = ws + o_x1;
  float* y1    = ws + o_y1;
  float* omax  = ws + o_omax;
  float* outw  = ws + o_outw;
  float* scr   = ws + o_scr;
  float* bdot  = ws + o_bdot;

  // ------------- bf16 workspace (element offsets, 8B-aligned) -------------
  bf16_t* wb = (bf16_t*)(ws + fo);
  long long bo = 0;
  auto balloc = [&](long long n) { long long o = bo; bo += (n + 3) & ~3LL; return o; };
  bf16_t* xbert_bf  = wb + balloc((long long)BB*LL*DIN);
  bf16_t* trans_bf  = wb + balloc((long long)BB*TT*LL);
  bf16_t* X_bf      = wb + balloc((long long)BB*TT*DIN);
  bf16_t* Wihf_bf   = wb + balloc((long long)DIN*G4);
  bf16_t* Wihb_bf   = wb + balloc((long long)DIN*G4);
  bf16_t* Whhf_bf   = wb + balloc((long long)HH*G4);
  bf16_t* Whhb_bf   = wb + balloc((long long)HH*G4);
  bf16_t* adj_bf    = wb + balloc((long long)BB*TT*TT);
  bf16_t* xcat_bf   = wb + balloc((long long)BB*TT*H2);
  bf16_t* h_bf      = wb + balloc(2*BB*HH);
  bf16_t* asp_bf    = wb + balloc(BB*H2);
  bf16_t* gA_bf     = wb + balloc(BB*H2);
  bf16_t* gB_bf     = wb + balloc(BB*H2);
  bf16_t* gc1w_bf   = wb + balloc((long long)H2*H2);
  bf16_t* gc2w_bf   = wb + balloc((long long)H2*H2);
  bf16_t* gwbf[2][3];
  for (int gi = 0; gi < 2; ++gi)
    for (int l = 0; l < 3; ++l) gwbf[gi][l] = wb + balloc((long long)H2*H2);
  bf16_t* hid_bf    = wb + balloc((long long)BB*TT*H2);
  bf16_t* gcn1_bf   = wb + balloc((long long)BB*TT*H2);
  bf16_t* cat2_bf   = wb + balloc((long long)BB*2*H2);
  bf16_t* fccat_bf  = wb + balloc((long long)BB*TT*2*H2);
  bf16_t* densew_bf = wb + balloc((long long)2*H2*PP);
  bf16_t* fcw_bf    = wb + balloc((long long)2*H2*PP);
  (void)balloc(64);   // tail pad: A K-tail overreads stay in-bounds

  auto conv = [&](const float* s, bf16_t* d, long long n) {
    unsigned blocks = (unsigned)((n + 255) / 256);
    to_bf16<<<blocks, 256, 0, stream>>>(s, d, n);
  };

  auto gemm = [&](const bf16_t* A, long long sA, int lda,
                  const bf16_t* Bp, long long sB, int ldb,
                  float* Cp, bf16_t* Cbf, long long sC, int ldc,
                  const float* bias,
                  const float* addm, long long sAdd, int ldadd,
                  const float* rdiv, long long sDiv,
                  const float* cscale, long long sScale,
                  int M, int N, int K, int nb, int act) {
    int tmB = ((M + 15) / 16 + 7) / 8;
    long long blocks = (long long)tmB * ((N + 31) / 32) * nb;
    gemm_wmma<<<(unsigned)blocks, 256, 0, stream>>>(
        A, sA, lda, Bp, sB, ldb, Cp, Cbf, sC, ldc,
        bias, addm, sAdd, ldadd, rdiv, sDiv, cscale, sScale, M, N, K, nb, act);
  };

  // ---- 0) stage constants + inputs to bf16; zero LSTM state ----
  conv(x_bert,    xbert_bf,  (long long)BB*LL*DIN);
  conv(transform, trans_bf,  (long long)BB*TT*LL);
  conv(Wih_f,     Wihf_bf,   (long long)DIN*G4);
  conv(Wih_b,     Wihb_bf,   (long long)DIN*G4);
  conv(Whh_f,     Whhf_bf,   (long long)HH*G4);
  conv(Whh_b,     Whhb_bf,   (long long)HH*G4);
  conv(adj,       adj_bf,    (long long)BB*TT*TT);
  conv(gc1_w,     gc1w_bf,   (long long)H2*H2);
  conv(gc2_w,     gc2w_bf,   (long long)H2*H2);
  for (int gi = 0; gi < 2; ++gi)
    for (int l = 0; l < 3; ++l) conv(gw[gi][l], gwbf[gi][l], (long long)H2*H2);
  conv(dense_w,   densew_bf, (long long)2*H2*PP);
  conv(fc_w,      fcw_bf,    (long long)2*H2*PP);
  zero_f32<<<(4*BB*HH + 255) / 256, 256, 0, stream>>>(state, 4*BB*HH);
  zero_bf<<<(2*BB*HH + 255) / 256, 256, 0, stream>>>(h_bf, 2*BB*HH);

  // ---- 1) X = transform @ x_bert  (bf16 out only; consumed by GEMMs) ----
  gemm(trans_bf, (long long)TT*LL, LL, xbert_bf, (long long)LL*DIN, DIN,
       nullptr, X_bf, (long long)TT*DIN, DIN, nullptr,
       nullptr, 0, 0, nullptr, 0, nullptr, 0, TT, DIN, LL, BB, 0);

  // ---- 2) pre = X @ Wih + b   (both directions) ----
  gemm(X_bf, 0, DIN, Wihf_bf, 0, G4, pre_f, nullptr, 0, G4, b_f,
       nullptr, 0, 0, nullptr, 0, nullptr, 0, BB*TT, G4, DIN, 1, 0);
  gemm(X_bf, 0, DIN, Wihb_bf, 0, G4, pre_b, nullptr, 0, G4, b_b,
       nullptr, 0, 0, nullptr, 0, nullptr, 0, BB*TT, G4, DIN, 1, 0);

  // ---- 3) LSTM recurrence, 128 steps, both directions ----
  for (int t = 0; t < TT; ++t) {
    gemm(h_bf, 0, HH, Whhf_bf, 0, G4, gts_f, nullptr, 0, G4, nullptr,
         pre_f + (long long)t * G4, 0, TT*G4, nullptr, 0, nullptr, 0,
         BB, G4, HH, 1, 0);
    gemm(h_bf + BB*HH, 0, HH, Whhb_bf, 0, G4, gts_b, nullptr, 0, G4, nullptr,
         pre_b + (long long)(TT - 1 - t) * G4, 0, TT*G4, nullptr, 0, nullptr, 0,
         BB, G4, HH, 1, 0);
    lstm_cell<<<(2*BB*HH + 255) / 256, 256, 0, stream>>>(state, h_bf, xcat, xcat_bf, t);
  }

  // ---- 4) aspect = xcat[b, anchor[b], :] ----
  aspect_gather<<<(BB*H2 + 255) / 256, 256, 0, stream>>>(xcat, anchor, asp, asp_bf);

  // ---- 5) gate MLPs (3 sigmoid layers each) -> g1, g2 ----
  float* gouts[2] = { g1, g2 };
  for (int gi = 0; gi < 2; ++gi) {
    gemm(asp_bf, 0, H2, gwbf[gi][0], 0, H2, nullptr, gA_bf, 0, H2, gb[gi][0],
         nullptr, 0, 0, nullptr, 0, nullptr, 0, BB, H2, H2, 1, 2);
    gemm(gA_bf, 0, H2, gwbf[gi][1], 0, H2, nullptr, gB_bf, 0, H2, gb[gi][1],
         nullptr, 0, 0, nullptr, 0, nullptr, 0, BB, H2, H2, 1, 2);
    gemm(gB_bf, 0, H2, gwbf[gi][2], 0, H2, gouts[gi], nullptr, 0, H2, gb[gi][2],
         nullptr, 0, 0, nullptr, 0, nullptr, 0, BB, H2, H2, 1, 2);
  }

  // ---- 6) GCN layer 1 ----
  row_degree<<<(BB*TT + 255) / 256, 256, 0, stream>>>(adj, den);
  gemm(xcat_bf, 0, H2, gc1w_bf, 0, H2, nullptr, hid_bf, 0, H2, nullptr,
       nullptr, 0, 0, nullptr, 0, nullptr, 0, BB*TT, H2, H2, 1, 0);
  gemm(adj_bf, (long long)TT*TT, TT, hid_bf, (long long)TT*H2, H2,
       gcn1, gcn1_bf, (long long)TT*H2, H2, gc1_b, nullptr, 0, 0,
       den, TT, nullptr, 0, TT, H2, TT, BB, 1);

  // ---- 7) x1/y1 maxima and xy scalar ----
  max_over_t<<<(BB*H2 + 255) / 256, 256, 0, stream>>>(gcn1, g1, x1);
  max_over_t<<<(BB*H2 + 255) / 256, 256, 0, stream>>>(gcn1, g2, y1);
  xy_reduce<<<1, 256, 0, stream>>>(x1, y1, out);

  // ---- 8) GCN layer 2 fused with g2 scaling -> x2 ----
  gemm(gcn1_bf, 0, H2, gc2w_bf, 0, H2, nullptr, hid_bf, 0, H2, nullptr,
       nullptr, 0, 0, nullptr, 0, nullptr, 0, BB*TT, H2, H2, 1, 0);
  gemm(adj_bf, (long long)TT*TT, TT, hid_bf, (long long)TT*H2, H2,
       x2, nullptr, (long long)TT*H2, H2, gc2_b, nullptr, 0, 0,
       den, TT, g2, H2, TT, H2, TT, BB, 1);

  // ---- 9) out = max_t x2 ; logits = [aspect, out] @ dense_w + dense_b ----
  max_over_t<<<(BB*H2 + 255) / 256, 256, 0, stream>>>(x2, nullptr, omax);
  concat2<<<(BB*2*H2 + 255) / 256, 256, 0, stream>>>(asp, omax, cat2_bf);
  gemm(cat2_bf, 0, 2*H2, densew_bf, 0, PP, out, nullptr, 0, PP, dense_b,
       nullptr, 0, 0, nullptr, 0, nullptr, 0, BB, PP, 2*H2, 1, 0);

  // ---- 10) output_w = [x2, aspect] @ fc_w + fc_b ; scores ; KL ----
  {
    long long n = (long long)BB * TT * 2 * H2;
    concat_fc<<<(unsigned)((n + 255) / 256), 256, 0, stream>>>(x2, asp, fccat_bf);
  }
  gemm(fccat_bf, 0, 2*H2, fcw_bf, 0, PP, outw, nullptr, 0, PP, fc_b,
       nullptr, 0, 0, nullptr, 0, nullptr, 0, BB*TT, PP, 2*H2, 1, 0);
  score_dot<<<(BB*TT + 255) / 256, 256, 0, stream>>>(out, outw, scr);
  kl_batch<<<BB, TT, 0, stream>>>(scr, dist, bdot);
  finalize_kl<<<1, 32, 0, stream>>>(bdot, out);
}
